// MoE_V3_Original_43946105372957
// MI455X (gfx1250) — matrix operand order repaired and checked by reading, not compiled, not run on hardware
//
#include <hip/hip_runtime.h>
#include <hip/hip_bf16.h>

// ---- problem constants (from reference) ----
#define NT     4096          // B*T tokens
#define CDIM   1024
#define FDIM   1024
#define FSDIM  2048
#define NEXP   8
#define NPAIR  (NT * 2)      // top-k = 2
#define SLOTS  NPAIR
#define BTILE  (128 * 32)    // B-tile elements per LDS buffer

// gfx1250 async global->LDS path (ASYNCcnt), with sync fallback
#if defined(__has_builtin)
#  if __has_builtin(__builtin_amdgcn_global_load_async_to_lds_b128)
#    define USE_ASYNC_LDS 1
#  endif
#endif
#ifndef USE_ASYNC_LDS
#  define USE_ASYNC_LDS 0
#endif

typedef __attribute__((ext_vector_type(16))) __bf16       v16bf;
typedef __attribute__((ext_vector_type(8)))  float        v8f;
typedef __attribute__((ext_vector_type(4)))  unsigned int v4u;
typedef __attribute__((ext_vector_type(4)))  int          v4i;

// address-space-typed vector pointers for the async builtin
typedef __attribute__((address_space(1))) v4i gv4i;   // global
typedef __attribute__((address_space(3))) v4i lv4i;   // LDS

struct FragBf { union { v16bf v; v4u q[2]; }; };

__device__ __forceinline__ unsigned short f2bf(float f) {
  union { float f; unsigned int u; } cv; cv.f = f;
  unsigned int u = cv.u;
  u += 0x7FFFu + ((u >> 16) & 1u);   // round-to-nearest-even
  return (unsigned short)(u >> 16);
}

__device__ __forceinline__ void wait_asynccnt0() {
#if defined(__has_builtin)
#  if __has_builtin(__builtin_amdgcn_s_wait_asynccnt)
  __builtin_amdgcn_s_wait_asynccnt(0);
#  else
  asm volatile("s_wait_asynccnt 0x0" ::: "memory");
#  endif
#else
  asm volatile("s_wait_asynccnt 0x0" ::: "memory");
#endif
}

// Stage one thread's 32-byte slice of a B tile into LDS.
__device__ __forceinline__ void stage_tile(const unsigned short* __restrict__ g,
                                           unsigned short* __restrict__ l) {
#if USE_ASYNC_LDS
  __builtin_amdgcn_global_load_async_to_lds_b128(
      (gv4i*)g, (lv4i*)l, 0, 0);
  __builtin_amdgcn_global_load_async_to_lds_b128(
      (gv4i*)(g + 8), (lv4i*)(l + 8), 0, 0);
#else
  v4u q0 = *(const v4u*)(g);
  v4u q1 = *(const v4u*)(g + 8);
  *(v4u*)(l)     = q0;
  *(v4u*)(l + 8) = q1;
#endif
}

// ---------------------------------------------------------------------------
// Block (8 waves) GEMM core. Each wave owns a 16(M) x 128(N) strip of the
// block's 128x128 tile. The 128x32 bf16 B tile is staged through LDS with
// double buffering: the NEXT tile's transfer (async global->LDS on gfx1250)
// is issued before the WMMA burst so it overlaps compute; the wave waits its
// own ASYNCcnt, then the block barrier publishes the tile.
//   aRow:   lane's A row, pre-offset by (lane>=16 ? 8 : 0) elements
//   wFetch: this thread's cooperative-fetch pointer:
//           W + (n0 + tid/2)*ldw + (tid&1)*16
//   smB:    LDS, 2 * 128*32 bf16 elements (16 KB)
// ---------------------------------------------------------------------------
__device__ __forceinline__ void gemm_core(
    const unsigned short* __restrict__ aRow,
    const unsigned short* __restrict__ wFetch,
    unsigned short* __restrict__ smB,
    int K, int lane, int tid, v8f acc[8])
{
  const int half = lane >> 4;
  const int nc   = lane & 15;
  const int fr   = tid >> 1;      // cooperative fetch: row within N-tile
  const int fh   = tid & 1;       // which 16-element half of the k-slice
  unsigned short* st0 = smB + fr * 32 + fh * 16;

  // prologue: stage tile k0=0 into buffer 0
  stage_tile(wFetch, st0);
#if USE_ASYNC_LDS
  wait_asynccnt0();
#endif
  __syncthreads();

  for (int k0 = 0; k0 < K; k0 += 32) {
    const int buf = (k0 >> 5) & 1;
    const unsigned short* smCur = smB + buf * BTILE;
    unsigned short*       smNxt = smB + (buf ^ 1) * BTILE;
    const bool more = (k0 + 32) < K;

    // A fragment for this k-step (global, branchless)
    FragBf a;
    a.q[0] = *(const v4u*)(aRow + k0);
    a.q[1] = *(const v4u*)(aRow + k0 + 16);

#if USE_ASYNC_LDS
    // async global->LDS for the NEXT tile; overlaps the WMMA burst below
    if (more) stage_tile(wFetch + k0 + 32, smNxt + fr * 32 + fh * 16);
#else
    v4u p0, p1;
    if (more) {
      p0 = *(const v4u*)(wFetch + k0 + 32);
      p1 = *(const v4u*)(wFetch + k0 + 40);
    }
#endif

    // read current B fragments from LDS
    FragBf b[8];
#pragma unroll
    for (int j = 0; j < 8; ++j) {
      const unsigned short* lp = smCur + (j * 16 + nc) * 32 + half * 8;
      b[j].q[0] = *(const v4u*)(lp);
      b[j].q[1] = *(const v4u*)(lp + 16);
    }

#pragma unroll
    for (int j = 0; j < 8; ++j)
      acc[j] = __builtin_amdgcn_wmma_f32_16x16x32_bf16(
          false, a.v, false, b[j].v, (short)0, acc[j], false, false);

#if USE_ASYNC_LDS
    wait_asynccnt0();
#else
    if (more) {
      unsigned short* sp = smNxt + fr * 32 + fh * 16;
      *(v4u*)(sp)     = p0;
      *(v4u*)(sp + 8) = p1;
    }
#endif
    __syncthreads();
  }
}

__device__ __forceinline__ void zero_acc(v8f acc[8]) {
#pragma unroll
  for (int j = 0; j < 8; ++j)
#pragma unroll
    for (int r = 0; r < 8; ++r) acc[j][r] = 0.f;
}

// ---------------------------------------------------------------------------
// fp32 -> bf16 conversion (vectorized x4; all sizes are multiples of 4)
// ---------------------------------------------------------------------------
__global__ void cvt_kernel(const float* __restrict__ s,
                           unsigned short* __restrict__ d, int n) {
  int i = (blockIdx.x * blockDim.x + threadIdx.x) * 4;
  if (i >= n) return;
  float4 f = *(const float4*)(s + i);
  d[i + 0] = f2bf(f.x); d[i + 1] = f2bf(f.y);
  d[i + 2] = f2bf(f.z); d[i + 3] = f2bf(f.w);
}

// ---------------------------------------------------------------------------
// Router: logits -> softmax -> top-2 -> renormalized weights
// ---------------------------------------------------------------------------
__global__ __launch_bounds__(256)
void router_kernel(const float* __restrict__ x, const float* __restrict__ wr,
                   int* __restrict__ top_idx, float* __restrict__ top_w) {
  int t = blockIdx.x * blockDim.x + threadIdx.x;
  if (t >= NT) return;
  const float* xp = x + (size_t)t * CDIM;
  float acc[NEXP];
#pragma unroll
  for (int e = 0; e < NEXP; ++e) acc[e] = 0.f;
  for (int c = 0; c < CDIM; c += 4) {
    float4 xv = *(const float4*)(xp + c);
#pragma unroll
    for (int e = 0; e < NEXP; ++e) {
      const float* w = wr + (size_t)e * CDIM + c;
      acc[e] += xv.x * w[0] + xv.y * w[1] + xv.z * w[2] + xv.w * w[3];
    }
  }
  float mx = acc[0];
#pragma unroll
  for (int e = 1; e < NEXP; ++e) mx = fmaxf(mx, acc[e]);
  float p[NEXP];
#pragma unroll
  for (int e = 0; e < NEXP; ++e) p[e] = __expf(acc[e] - mx);
  int i1 = 0;
#pragma unroll
  for (int e = 1; e < NEXP; ++e) if (p[e] > p[i1]) i1 = e;
  int i2 = (i1 == 0) ? 1 : 0;
#pragma unroll
  for (int e = 0; e < NEXP; ++e) if (e != i1 && p[e] > p[i2]) i2 = e;
  float s = p[i1] + p[i2];
  top_idx[t * 2]     = i1;
  top_idx[t * 2 + 1] = i2;
  top_w[t * 2]       = p[i1] / s;
  top_w[t * 2 + 1]   = p[i2] / s;
}

// ---------------------------------------------------------------------------
// Deterministic dispatch: histogram, scan, stable rank, scatter
// ---------------------------------------------------------------------------
__global__ void hist_kernel(const int* __restrict__ top_idx, int* __restrict__ cnt) {
  int i = blockIdx.x * blockDim.x + threadIdx.x;
  if (i < NPAIR) atomicAdd(&cnt[top_idx[i]], 1);
}

__global__ void scan_kernel(const int* __restrict__ cnt, int* __restrict__ offs) {
  if (threadIdx.x == 0 && blockIdx.x == 0) {
    int s = 0;
    for (int e = 0; e < NEXP; ++e) { offs[e] = s; s += cnt[e]; }
    offs[NEXP] = s;
  }
}

__global__ void rank_kernel(const int* __restrict__ top_idx, int* __restrict__ pair_pos) {
  int i = blockIdx.x * blockDim.x + threadIdx.x;
  if (i >= NPAIR) return;
  int e = top_idx[i];
  int r = 0;
  for (int j = 0; j < i; ++j) r += (top_idx[j] == e) ? 1 : 0;
  pair_pos[i] = r;   // stable position within expert bin
}

__global__ void scatter_kernel(const int* __restrict__ top_idx,
                               const float* __restrict__ top_w,
                               const int* __restrict__ offs,
                               const int* __restrict__ pair_pos,
                               int* __restrict__ slot_tok,
                               float* __restrict__ slot_w,
                               int* __restrict__ pair_slot) {
  int i = blockIdx.x * blockDim.x + threadIdx.x;
  if (i >= NPAIR) return;
  int e = top_idx[i];
  int pos = offs[e] + pair_pos[i];
  slot_tok[pos] = i >> 1;
  slot_w[pos]   = top_w[i];
  pair_slot[i]  = pos;
}

// ---------------------------------------------------------------------------
// Expert GEMM1: inter[slot, F] = sqrelu( x[tok(slot)] @ w1[e]^T )   (bf16 out)
// ---------------------------------------------------------------------------
__global__ __launch_bounds__(256)
void expert_gemm1_kernel(const unsigned short* __restrict__ xb,
                         const unsigned short* __restrict__ w1b,
                         unsigned short* __restrict__ interE,
                         const int* __restrict__ slot_tok,
                         const int* __restrict__ offs,
                         const int* __restrict__ cnt,
                         const unsigned short* __restrict__ zrow) {
  __shared__ unsigned short smB[2 * BTILE];
  const int e = blockIdx.z;
  const int cntE = cnt[e];
  const int m0blk = blockIdx.y * 128;
  if (m0blk >= cntE) return;
  const int n0   = blockIdx.x * 128;
  const int tid  = threadIdx.x;
  const int wave = tid >> 5;
  const int lane = tid & 31;
  const int half = lane >> 4;
  const int nc   = lane & 15;
  const int m0   = m0blk + wave * 16;
  const int base = offs[e];

  const int arow = m0 + nc;
  const int tok  = (arow < cntE) ? slot_tok[base + arow] : 0;
  const unsigned short* aRow =
      (arow < cntE) ? xb + (size_t)tok * CDIM + half * 8 : zrow + half * 8;

  const unsigned short* wBase = w1b + (size_t)e * FDIM * CDIM;
  const unsigned short* wFetch =
      wBase + (size_t)(n0 + (tid >> 1)) * CDIM + (tid & 1) * 16;

  v8f acc[8];
  zero_acc(acc);
  gemm_core(aRow, wFetch, smB, CDIM, lane, tid, acc);

#pragma unroll
  for (int j = 0; j < 8; ++j)
#pragma unroll
    for (int r = 0; r < 8; ++r) {
      int lr = m0 + r + half * 8;
      if (lr < cntE) {
        float v = acc[j][r];
        v = v > 0.f ? v * v : 0.f;
        interE[(size_t)(base + lr) * FDIM + n0 + j * 16 + nc] = f2bf(v);
      }
    }
}

// ---------------------------------------------------------------------------
// Expert GEMM2: comb[slot, C] = w(slot) * ( inter[slot] @ w2[e]^T )  (f32 out)
// ---------------------------------------------------------------------------
__global__ __launch_bounds__(256)
void expert_gemm2_kernel(const unsigned short* __restrict__ interE,
                         const unsigned short* __restrict__ w2b,
                         float* __restrict__ combE,
                         const float* __restrict__ slot_w,
                         const int* __restrict__ offs,
                         const int* __restrict__ cnt,
                         const unsigned short* __restrict__ zrow) {
  __shared__ unsigned short smB[2 * BTILE];
  const int e = blockIdx.z;
  const int cntE = cnt[e];
  const int m0blk = blockIdx.y * 128;
  if (m0blk >= cntE) return;
  const int n0   = blockIdx.x * 128;
  const int tid  = threadIdx.x;
  const int wave = tid >> 5;
  const int lane = tid & 31;
  const int half = lane >> 4;
  const int nc   = lane & 15;
  const int m0   = m0blk + wave * 16;
  const int base = offs[e];

  const int arow = m0 + nc;
  const unsigned short* aRow =
      (arow < cntE) ? interE + (size_t)(base + arow) * FDIM + half * 8
                    : zrow + half * 8;

  const unsigned short* wBase = w2b + (size_t)e * CDIM * FDIM;
  const unsigned short* wFetch =
      wBase + (size_t)(n0 + (tid >> 1)) * FDIM + (tid & 1) * 16;

  v8f acc[8];
  zero_acc(acc);
  gemm_core(aRow, wFetch, smB, FDIM, lane, tid, acc);

#pragma unroll
  for (int j = 0; j < 8; ++j)
#pragma unroll
    for (int r = 0; r < 8; ++r) {
      int lr = m0 + r + half * 8;
      if (lr < cntE) {
        int slot = base + lr;
        combE[(size_t)slot * CDIM + n0 + j * 16 + nc] = slot_w[slot] * acc[j][r];
      }
    }
}

// ---------------------------------------------------------------------------
// Shared expert GEMM1: interS = sqrelu( x @ ws1^T )   (bf16 out)
// ---------------------------------------------------------------------------
__global__ __launch_bounds__(256)
void shared_gemm1_kernel(const unsigned short* __restrict__ xb,
                         const unsigned short* __restrict__ ws1b,
                         unsigned short* __restrict__ interS) {
  __shared__ unsigned short smB[2 * BTILE];
  const int n0   = blockIdx.x * 128;
  const int tid  = threadIdx.x;
  const int wave = tid >> 5;
  const int lane = tid & 31;
  const int half = lane >> 4;
  const int nc   = lane & 15;
  const int m0   = blockIdx.y * 128 + wave * 16;

  const unsigned short* aRow = xb + (size_t)(m0 + nc) * CDIM + half * 8;
  const unsigned short* wFetch =
      ws1b + (size_t)(n0 + (tid >> 1)) * CDIM + (tid & 1) * 16;

  v8f acc[8];
  zero_acc(acc);
  gemm_core(aRow, wFetch, smB, CDIM, lane, tid, acc);

#pragma unroll
  for (int j = 0; j < 8; ++j)
#pragma unroll
    for (int r = 0; r < 8; ++r) {
      int row = m0 + r + half * 8;
      float v = acc[j][r];
      v = v > 0.f ? v * v : 0.f;
      interS[(size_t)row * FSDIM + n0 + j * 16 + nc] = f2bf(v);
    }
}

// ---------------------------------------------------------------------------
// Shared expert GEMM2 + MoE combine:
//   out[t, c] = interS[t] @ ws2^T  +  comb[slot(t,0), c] + comb[slot(t,1), c]
// Each output element written exactly once (deterministic, no atomics).
// ---------------------------------------------------------------------------
__global__ __launch_bounds__(256)
void shared_gemm2_kernel(const unsigned short* __restrict__ interS,
                         const unsigned short* __restrict__ ws2b,
                         const float* __restrict__ combE,
                         const int* __restrict__ pair_slot,
                         float* __restrict__ out) {
  __shared__ unsigned short smB[2 * BTILE];
  const int n0   = blockIdx.x * 128;
  const int tid  = threadIdx.x;
  const int wave = tid >> 5;
  const int lane = tid & 31;
  const int half = lane >> 4;
  const int nc   = lane & 15;
  const int m0   = blockIdx.y * 128 + wave * 16;

  const unsigned short* aRow = interS + (size_t)(m0 + nc) * FSDIM + half * 8;
  const unsigned short* wFetch =
      ws2b + (size_t)(n0 + (tid >> 1)) * FSDIM + (tid & 1) * 16;

  v8f acc[8];
  zero_acc(acc);
  gemm_core(aRow, wFetch, smB, FSDIM, lane, tid, acc);

#pragma unroll
  for (int r = 0; r < 8; ++r) {
    const int row = m0 + r + half * 8;
    const int s0 = pair_slot[2 * row];
    const int s1 = pair_slot[2 * row + 1];
    const float* c0 = combE + (size_t)s0 * CDIM;
    const float* c1 = combE + (size_t)s1 * CDIM;
#pragma unroll
    for (int j = 0; j < 8; ++j) {
      const int col = n0 + j * 16 + nc;
      out[(size_t)row * CDIM + col] = acc[j][r] + c0[col] + c1[col];
    }
  }
}

// ---------------------------------------------------------------------------
extern "C" void kernel_launch(void* const* d_in, const int* in_sizes, int n_in,
                              void* d_out, int out_size, void* d_ws, size_t ws_size,
                              hipStream_t stream) {
  (void)in_sizes; (void)n_in; (void)out_size; (void)ws_size;
  const float* x   = (const float*)d_in[0];
  const float* wr  = (const float*)d_in[1];
  const float* w1  = (const float*)d_in[2];
  const float* w2  = (const float*)d_in[3];
  const float* ws1 = (const float*)d_in[4];
  const float* ws2 = (const float*)d_in[5];
  float* out = (float*)d_out;

  char* base = (char*)d_ws;
  size_t cur = 0;
  auto alloc = [&](size_t bytes) -> void* {
    void* p = base + cur;
    cur += (bytes + 255) & ~(size_t)255;
    return p;
  };
  unsigned short* xb     = (unsigned short*)alloc((size_t)NT * CDIM * 2);
  unsigned short* w1b    = (unsigned short*)alloc((size_t)NEXP * FDIM * CDIM * 2);
  unsigned short* w2b    = (unsigned short*)alloc((size_t)NEXP * CDIM * FDIM * 2);
  unsigned short* ws1b   = (unsigned short*)alloc((size_t)FSDIM * CDIM * 2);
  unsigned short* ws2b   = (unsigned short*)alloc((size_t)CDIM * FSDIM * 2);
  unsigned short* interE = (unsigned short*)alloc((size_t)SLOTS * FDIM * 2);
  unsigned short* interS = (unsigned short*)alloc((size_t)NT * FSDIM * 2);
  float* combE     = (float*)alloc((size_t)SLOTS * CDIM * 4);
  int*   top_idx   = (int*)alloc((size_t)NPAIR * 4);
  float* top_w     = (float*)alloc((size_t)NPAIR * 4);
  int*   cnt       = (int*)alloc((size_t)NEXP * 4);
  int*   offs      = (int*)alloc((size_t)(NEXP + 1) * 4);
  int*   pair_pos  = (int*)alloc((size_t)NPAIR * 4);
  int*   slot_tok  = (int*)alloc((size_t)SLOTS * 4);
  float* slot_w    = (float*)alloc((size_t)SLOTS * 4);
  int*   pair_slot = (int*)alloc((size_t)NPAIR * 4);
  unsigned short* zrow = (unsigned short*)alloc((size_t)FSDIM * 2); // zero dummy row

  (void)hipMemsetAsync(cnt, 0, NEXP * sizeof(int), stream);
  (void)hipMemsetAsync(zrow, 0, (size_t)FSDIM * 2, stream);

  // fp32 -> bf16 conversions
  {
    int n;
    n = NT * CDIM;          cvt_kernel<<<n / 4 / 256, 256, 0, stream>>>(x,   xb,   n);
    n = NEXP * FDIM * CDIM; cvt_kernel<<<n / 4 / 256, 256, 0, stream>>>(w1,  w1b,  n);
    n = NEXP * CDIM * FDIM; cvt_kernel<<<n / 4 / 256, 256, 0, stream>>>(w2,  w2b,  n);
    n = FSDIM * CDIM;       cvt_kernel<<<n / 4 / 256, 256, 0, stream>>>(ws1, ws1b, n);
    n = CDIM * FSDIM;       cvt_kernel<<<n / 4 / 256, 256, 0, stream>>>(ws2, ws2b, n);
  }

  router_kernel<<<NT / 256, 256, 0, stream>>>(x, wr, top_idx, top_w);
  hist_kernel<<<NPAIR / 256, 256, 0, stream>>>(top_idx, cnt);
  scan_kernel<<<1, 32, 0, stream>>>(cnt, offs);
  rank_kernel<<<NPAIR / 256, 256, 0, stream>>>(top_idx, pair_pos);
  scatter_kernel<<<NPAIR / 256, 256, 0, stream>>>(top_idx, top_w, offs, pair_pos,
                                                  slot_tok, slot_w, pair_slot);

  dim3 blk(256);
  expert_gemm1_kernel<<<dim3(FDIM / 128, NT / 128, NEXP), blk, 0, stream>>>(
      xb, w1b, interE, slot_tok, offs, cnt, zrow);
  expert_gemm2_kernel<<<dim3(CDIM / 128, NT / 128, NEXP), blk, 0, stream>>>(
      interE, w2b, combE, slot_w, offs, cnt, zrow);
  shared_gemm1_kernel<<<dim3(FSDIM / 128, NT / 128), blk, 0, stream>>>(xb, ws1b, interS);
  shared_gemm2_kernel<<<dim3(CDIM / 128, NT / 128), blk, 0, stream>>>(
      interS, ws2b, combE, pair_slot, out);
}
